// InfoGraph_68289980006827
// MI455X (gfx1250) — matrix operand order repaired and checked by reading, not compiled, not run on hardware
//
#include <hip/hip_runtime.h>
#include <hip/hip_bf16.h>

typedef _Float16 v16h __attribute__((ext_vector_type(16)));
typedef _Float16 v8h  __attribute__((ext_vector_type(8)));
typedef float    v8f  __attribute__((ext_vector_type(8)));

#define G_DIM 64  // feature dim everywhere

__device__ inline v8f wmma16(v16h a, v16h b, v8f c) {
  // D = A(16x32 f16) * B(32x16 f16) + C(16x16 f32)
  return __builtin_amdgcn_wmma_f32_16x16x32_f16(false, a, false, b, (short)0, c, false, false);
}

// ---- fragment loaders (wave32 layouts per CDNA5 ISA 7.12.2) ----

// A fragment for k-chunk kc (0/1) from f32 row-major [*,64]: val = alpha*p + q
__device__ inline v16h load_a_combine(const float* p, const float* q, float alpha,
                                      int row, int valid, int lane, int kc) {
  int kb = ((lane >> 4) << 3) + (kc << 5);   // lanes>=16 start at K=8; kc adds 32
  v16h a;
  if (valid) {
    const float* pr = p + (long long)row * G_DIM + kb;
    const float* qr = q + (long long)row * G_DIM + kb;
    float4 p0 = *(const float4*)(pr);      float4 q0 = *(const float4*)(qr);
    float4 p1 = *(const float4*)(pr + 4);  float4 q1 = *(const float4*)(qr + 4);
    float4 p2 = *(const float4*)(pr + 16); float4 q2 = *(const float4*)(qr + 16);
    float4 p3 = *(const float4*)(pr + 20); float4 q3 = *(const float4*)(qr + 20);
    a[0]  = (_Float16)(alpha * p0.x + q0.x); a[1]  = (_Float16)(alpha * p0.y + q0.y);
    a[2]  = (_Float16)(alpha * p0.z + q0.z); a[3]  = (_Float16)(alpha * p0.w + q0.w);
    a[4]  = (_Float16)(alpha * p1.x + q1.x); a[5]  = (_Float16)(alpha * p1.y + q1.y);
    a[6]  = (_Float16)(alpha * p1.z + q1.z); a[7]  = (_Float16)(alpha * p1.w + q1.w);
    a[8]  = (_Float16)(alpha * p2.x + q2.x); a[9]  = (_Float16)(alpha * p2.y + q2.y);
    a[10] = (_Float16)(alpha * p2.z + q2.z); a[11] = (_Float16)(alpha * p2.w + q2.w);
    a[12] = (_Float16)(alpha * p3.x + q3.x); a[13] = (_Float16)(alpha * p3.y + q3.y);
    a[14] = (_Float16)(alpha * p3.z + q3.z); a[15] = (_Float16)(alpha * p3.w + q3.w);
  } else {
#pragma unroll
    for (int j = 0; j < 16; ++j) a[j] = (_Float16)0.f;
  }
  return a;
}

// A fragment from f32 sums scaled by 1/max(counts,1)
__device__ inline v16h load_a_mean(const float* sums, const float* counts,
                                   int row, int valid, int lane, int kc) {
  int kb = ((lane >> 4) << 3) + (kc << 5);
  v16h a;
  if (valid) {
    float inv = 1.0f / fmaxf(counts[row], 1.0f);
    const float* pr = sums + (long long)row * G_DIM + kb;
    float4 p0 = *(const float4*)(pr);
    float4 p1 = *(const float4*)(pr + 4);
    float4 p2 = *(const float4*)(pr + 16);
    float4 p3 = *(const float4*)(pr + 20);
    a[0]  = (_Float16)(p0.x * inv); a[1]  = (_Float16)(p0.y * inv);
    a[2]  = (_Float16)(p0.z * inv); a[3]  = (_Float16)(p0.w * inv);
    a[4]  = (_Float16)(p1.x * inv); a[5]  = (_Float16)(p1.y * inv);
    a[6]  = (_Float16)(p1.z * inv); a[7]  = (_Float16)(p1.w * inv);
    a[8]  = (_Float16)(p2.x * inv); a[9]  = (_Float16)(p2.y * inv);
    a[10] = (_Float16)(p2.z * inv); a[11] = (_Float16)(p2.w * inv);
    a[12] = (_Float16)(p3.x * inv); a[13] = (_Float16)(p3.y * inv);
    a[14] = (_Float16)(p3.z * inv); a[15] = (_Float16)(p3.w * inv);
  } else {
#pragma unroll
    for (int j = 0; j < 16; ++j) a[j] = (_Float16)0.f;
  }
  return a;
}

// A fragment from per-wave f16 LDS tile [16][64] (row = M, col = K).
// The two 8-half runs are contiguous & 16B aligned -> 2x ds_load_b128.
__device__ inline v16h load_a_lds(const _Float16* act, int lane, int kc) {
  int row = lane & 15;
  int kb = ((lane >> 4) << 3) + (kc << 5);
  const _Float16* p = act + row * G_DIM + kb;
  v8h lo = *(const v8h*)(p);        // K = kb .. kb+7
  v8h hi = *(const v8h*)(p + 16);   // K = kb+16 .. kb+23
  return __builtin_shufflevector(lo, hi, 0, 1, 2, 3, 4, 5, 6, 7,
                                 8, 9, 10, 11, 12, 13, 14, 15);
}

// B fragment from fragment-major LDS: sW[((kc*4+nt)*32+lane)*16 + t].
// One aligned 32B load per lane -> 2x ds_load_b128, zero packing ALU.
__device__ inline v16h load_b_frag(const _Float16* Wf, int lane, int nt, int kc) {
  const _Float16* p = Wf + ((((kc << 2) | nt) << 5) | lane) * 16;
  return *(const v16h*)p;
}

// Swizzled fill of a 64x64 f16 weight matrix into fragment-major LDS layout.
// dest index i = frag*512 + lane*16 + t ; frag = kc*4+nt
__device__ inline void fill_b_frags(_Float16* Wf, const float* W, int tid) {
  for (int i = tid; i < 4096; i += 256) {
    int t    = i & 15;
    int lane = (i >> 4) & 31;
    int frag = i >> 9;
    int kc = frag >> 2, nt = frag & 3;
    int col = (lane & 15) + nt * 16;
    int K   = (kc << 5) + ((lane >> 4) << 3) + (t & 7) + ((t >> 3) << 4);
    Wf[i] = (_Float16)W[K * G_DIM + col];
  }
}

// ---------------- utility kernels ----------------

__global__ void zero_kernel(float* p, long long n) {
  long long i = (long long)blockIdx.x * blockDim.x + threadIdx.x;
  if (i < n) p[i] = 0.f;
}

__global__ void scatter_edges_kernel(const float* __restrict__ feat,
                                     const int* __restrict__ src,
                                     const int* __restrict__ dst,
                                     float* agg, int E) {
  long long i = (long long)blockIdx.x * blockDim.x + threadIdx.x;
  if (i >= (long long)E * 16) return;
  int e = (int)(i >> 4);
  int c = ((int)i & 15) * 4;
  const float4 v = *(const float4*)(feat + (long long)src[e] * G_DIM + c);
  float* d = agg + (long long)dst[e] * G_DIM + c;
  __hip_atomic_fetch_add(d + 0, v.x, __ATOMIC_RELAXED, __HIP_MEMORY_SCOPE_AGENT);
  __hip_atomic_fetch_add(d + 1, v.y, __ATOMIC_RELAXED, __HIP_MEMORY_SCOPE_AGENT);
  __hip_atomic_fetch_add(d + 2, v.z, __ATOMIC_RELAXED, __HIP_MEMORY_SCOPE_AGENT);
  __hip_atomic_fetch_add(d + 3, v.w, __ATOMIC_RELAXED, __HIP_MEMORY_SCOPE_AGENT);
}

__global__ void graph_accum_kernel(const float* __restrict__ h,
                                   const int* __restrict__ gids,
                                   float* sums, float* counts, int N) {
  long long i = (long long)blockIdx.x * blockDim.x + threadIdx.x;
  if (i >= (long long)N * 16) return;
  int n = (int)(i >> 4);
  int c = ((int)i & 15) * 4;
  int g = gids[n];
  const float4 v = *(const float4*)(h + (long long)n * G_DIM + c);
  float* s = sums + (long long)g * G_DIM + c;
  __hip_atomic_fetch_add(s + 0, v.x, __ATOMIC_RELAXED, __HIP_MEMORY_SCOPE_AGENT);
  __hip_atomic_fetch_add(s + 1, v.y, __ATOMIC_RELAXED, __HIP_MEMORY_SCOPE_AGENT);
  __hip_atomic_fetch_add(s + 2, v.z, __ATOMIC_RELAXED, __HIP_MEMORY_SCOPE_AGENT);
  __hip_atomic_fetch_add(s + 3, v.w, __ATOMIC_RELAXED, __HIP_MEMORY_SCOPE_AGENT);
  if (c == 0)
    __hip_atomic_fetch_add(&counts[g], 1.0f, __ATOMIC_RELAXED, __HIP_MEMORY_SCOPE_AGENT);
}

// ---------------- GIN layer 0: two-layer MLP + BN, WMMA ----------------

__global__ __launch_bounds__(256) void gin_mlp0_kernel(
    const float* __restrict__ x, const float* __restrict__ agg, const float* __restrict__ eps0,
    const float* __restrict__ W0a, const float* __restrict__ bb0a,
    const float* __restrict__ W0b, const float* __restrict__ bb0b,
    const float* __restrict__ g0i, const float* __restrict__ be0i,
    const float* __restrict__ m0i, const float* __restrict__ v0i,
    const float* __restrict__ g0,  const float* __restrict__ be0,
    const float* __restrict__ m0,  const float* __restrict__ v0,
    float* __restrict__ h_out, int N) {
  __shared__ __attribute__((aligned(32))) _Float16 sWa[64 * 64];
  __shared__ __attribute__((aligned(32))) _Float16 sWb[64 * 64];
  __shared__ __attribute__((aligned(32))) _Float16 sAct[8][16 * 64];

  int tid = threadIdx.x;
  fill_b_frags(sWa, W0a, tid);
  fill_b_frags(sWb, W0b, tid);
  __syncthreads();

  int wave = tid >> 5, lane = tid & 31;
  int node_base = blockIdx.x * 128 + wave * 16;
  int col = lane & 15;
  int mb = (lane >> 4) * 8;
  int arow = node_base + (lane & 15);
  int avalid = arow < N;
  float alpha = 1.0f + eps0[0];

  v16h a0 = load_a_combine(x, agg, alpha, arow, avalid, lane, 0);
  v16h a1 = load_a_combine(x, agg, alpha, arow, avalid, lane, 1);

  v8f acc[4];
#pragma unroll
  for (int nt = 0; nt < 4; ++nt) {
    v8f c = {};
    c = wmma16(a0, load_b_frag(sWa, lane, nt, 0), c);
    c = wmma16(a1, load_b_frag(sWa, lane, nt, 1), c);
    acc[nt] = c;
  }

  _Float16* act = &sAct[wave][0];
#pragma unroll
  for (int nt = 0; nt < 4; ++nt) {
    float bias = bb0a[nt * 16 + col];
#pragma unroll
    for (int r = 0; r < 8; ++r) {
      float v = acc[nt][r] + bias;
      act[(mb + r) * G_DIM + nt * 16 + col] = (_Float16)fmaxf(v, 0.f);
    }
  }
  __syncthreads();

  v16h c0 = load_a_lds(act, lane, 0);
  v16h c1 = load_a_lds(act, lane, 1);
#pragma unroll
  for (int nt = 0; nt < 4; ++nt) {
    v8f c = {};
    c = wmma16(c0, load_b_frag(sWb, lane, nt, 0), c);
    c = wmma16(c1, load_b_frag(sWb, lane, nt, 1), c);
    acc[nt] = c;
  }

#pragma unroll
  for (int nt = 0; nt < 4; ++nt) {
    int n = nt * 16 + col;
    float si = g0i[n] * rsqrtf(v0i[n] + 1e-5f);
    float ti = be0i[n] - m0i[n] * si;
    float so = g0[n] * rsqrtf(v0[n] + 1e-5f);
    float to = be0[n] - m0[n] * so;
    float S = si * so, T = ti * so + to;
    float bias = bb0b[n];
#pragma unroll
    for (int r = 0; r < 8; ++r) {
      int row = node_base + mb + r;
      if (row < N) {
        float v = fmaxf(acc[nt][r] + bias, 0.f);   // relu(W0b + b)
        h_out[(long long)row * G_DIM + n] = fmaxf(v * S + T, 0.f);  // bn_i -> relu(bn_o)
      }
    }
  }
}

// ---------------- GIN layer 1: single MLP + BN, in-place ----------------

__global__ __launch_bounds__(256) void gin_mlp1_kernel(
    const float* h_in, const float* __restrict__ agg, const float* __restrict__ eps1,
    const float* __restrict__ W1a, const float* __restrict__ bb1a,
    const float* __restrict__ g1i, const float* __restrict__ be1i,
    const float* __restrict__ m1i, const float* __restrict__ v1i,
    const float* __restrict__ g1,  const float* __restrict__ be1,
    const float* __restrict__ m1,  const float* __restrict__ v1,
    float* h_out, int N) {
  __shared__ __attribute__((aligned(32))) _Float16 sW[64 * 64];
  int tid = threadIdx.x;
  fill_b_frags(sW, W1a, tid);
  __syncthreads();

  int wave = tid >> 5, lane = tid & 31;
  int node_base = blockIdx.x * 128 + wave * 16;
  int col = lane & 15;
  int mb = (lane >> 4) * 8;
  int arow = node_base + (lane & 15);
  int avalid = arow < N;
  float alpha = 1.0f + eps1[0];

  v16h a0 = load_a_combine(h_in, agg, alpha, arow, avalid, lane, 0);
  v16h a1 = load_a_combine(h_in, agg, alpha, arow, avalid, lane, 1);

  v8f acc[4];
#pragma unroll
  for (int nt = 0; nt < 4; ++nt) {
    v8f c = {};
    c = wmma16(a0, load_b_frag(sW, lane, nt, 0), c);
    c = wmma16(a1, load_b_frag(sW, lane, nt, 1), c);
    acc[nt] = c;
  }

#pragma unroll
  for (int nt = 0; nt < 4; ++nt) {
    int n = nt * 16 + col;
    float si = g1i[n] * rsqrtf(v1i[n] + 1e-5f);
    float ti = be1i[n] - m1i[n] * si;
    float so = g1[n] * rsqrtf(v1[n] + 1e-5f);
    float to = be1[n] - m1[n] * so;
    float S = si * so, T = ti * so + to;
    float bias = bb1a[n];
#pragma unroll
    for (int r = 0; r < 8; ++r) {
      int row = node_base + mb + r;
      if (row < N) {
        float v = fmaxf(acc[nt][r] + bias, 0.f);
        h_out[(long long)row * G_DIM + n] = fmaxf(v * S + T, 0.f);
      }
    }
  }
}

// ---------------- readout MLP on per-graph means ----------------

__global__ __launch_bounds__(256) void readout_kernel(
    const float* __restrict__ sums, const float* __restrict__ counts,
    const float* __restrict__ Wr1, const float* __restrict__ br1,
    const float* __restrict__ Wr2, const float* __restrict__ br2,
    float* __restrict__ out, int G) {
  __shared__ __attribute__((aligned(32))) _Float16 sW1[64 * 64];
  __shared__ __attribute__((aligned(32))) _Float16 sW2[64 * 64];
  __shared__ __attribute__((aligned(32))) _Float16 sAct[8][16 * 64];

  int tid = threadIdx.x;
  fill_b_frags(sW1, Wr1, tid);
  fill_b_frags(sW2, Wr2, tid);
  __syncthreads();

  int wave = tid >> 5, lane = tid & 31;
  int row_base = blockIdx.x * 128 + wave * 16;
  int col = lane & 15;
  int mb = (lane >> 4) * 8;
  int arow = row_base + (lane & 15);
  int avalid = arow < G;

  v16h a0 = load_a_mean(sums, counts, arow, avalid, lane, 0);
  v16h a1 = load_a_mean(sums, counts, arow, avalid, lane, 1);

  v8f acc[4];
#pragma unroll
  for (int nt = 0; nt < 4; ++nt) {
    v8f c = {};
    c = wmma16(a0, load_b_frag(sW1, lane, nt, 0), c);
    c = wmma16(a1, load_b_frag(sW1, lane, nt, 1), c);
    acc[nt] = c;
  }

  _Float16* act = &sAct[wave][0];
#pragma unroll
  for (int nt = 0; nt < 4; ++nt) {
    float bias = br1[nt * 16 + col];
#pragma unroll
    for (int r = 0; r < 8; ++r)
      act[(mb + r) * G_DIM + nt * 16 + col] = (_Float16)fmaxf(acc[nt][r] + bias, 0.f);
  }
  __syncthreads();

  v16h c0 = load_a_lds(act, lane, 0);
  v16h c1 = load_a_lds(act, lane, 1);
#pragma unroll
  for (int nt = 0; nt < 4; ++nt) {
    v8f c = {};
    c = wmma16(c0, load_b_frag(sW2, lane, nt, 0), c);
    c = wmma16(c1, load_b_frag(sW2, lane, nt, 1), c);
    acc[nt] = c;
  }

#pragma unroll
  for (int nt = 0; nt < 4; ++nt) {
    int n = nt * 16 + col;
    float bias = br2[n];
#pragma unroll
    for (int r = 0; r < 8; ++r) {
      int row = row_base + mb + r;
      if (row < G) out[(long long)row * G_DIM + n] = acc[nt][r] + bias;
    }
  }
}

// ---------------- host-side launch ----------------

extern "C" void kernel_launch(void* const* d_in, const int* in_sizes, int n_in,
                              void* d_out, int out_size, void* d_ws, size_t ws_size,
                              hipStream_t stream) {
  const float* x    = (const float*)d_in[0];
  const int*   src  = (const int*)d_in[1];
  const int*   dst  = (const int*)d_in[2];
  const int*   gids = (const int*)d_in[3];
  const float* eps0 = (const float*)d_in[4];
  const float* W0a  = (const float*)d_in[5];
  const float* bb0a = (const float*)d_in[6];
  const float* W0b  = (const float*)d_in[7];
  const float* bb0b = (const float*)d_in[8];
  const float* g0i  = (const float*)d_in[9];
  const float* be0i = (const float*)d_in[10];
  const float* m0i  = (const float*)d_in[11];
  const float* v0i  = (const float*)d_in[12];
  const float* g0   = (const float*)d_in[13];
  const float* be0  = (const float*)d_in[14];
  const float* m0   = (const float*)d_in[15];
  const float* v0   = (const float*)d_in[16];
  const float* eps1 = (const float*)d_in[17];
  const float* W1a  = (const float*)d_in[18];
  const float* bb1a = (const float*)d_in[19];
  const float* g1i  = (const float*)d_in[20];
  const float* be1i = (const float*)d_in[21];
  const float* m1i  = (const float*)d_in[22];
  const float* v1i  = (const float*)d_in[23];
  const float* g1   = (const float*)d_in[24];
  const float* be1  = (const float*)d_in[25];
  const float* m1   = (const float*)d_in[26];
  const float* v1   = (const float*)d_in[27];
  const float* Wr1  = (const float*)d_in[28];
  const float* br1  = (const float*)d_in[29];
  const float* Wr2  = (const float*)d_in[30];
  const float* br2  = (const float*)d_in[31];

  const int N = in_sizes[0] / G_DIM;             // 50000
  const int E = in_sizes[1];                     // 800000
  const int G = (out_size - N * G_DIM) / G_DIM;  // 500

  float* out  = (float*)d_out;
  float* hbuf = out + (long long)G * G_DIM;       // node features region of d_out
  float* ws     = (float*)d_ws;
  float* agg    = ws;                              // N*64
  float* sums   = ws + (long long)N * G_DIM;       // G*64
  float* counts = sums + (long long)G * G_DIM;     // G

  const int TB = 256;
  const long long aggN = (long long)N * G_DIM;
  const long long ztot = aggN + (long long)G * G_DIM + G;  // agg + sums + counts

  zero_kernel<<<(unsigned)((ztot + TB - 1) / TB), TB, 0, stream>>>(ws, ztot);

  scatter_edges_kernel<<<(unsigned)(((long long)E * 16 + TB - 1) / TB), TB, 0, stream>>>(
      x, src, dst, agg, E);

  gin_mlp0_kernel<<<(unsigned)((N + 127) / 128), TB, 0, stream>>>(
      x, agg, eps0, W0a, bb0a, W0b, bb0b,
      g0i, be0i, m0i, v0i, g0, be0, m0, v0, hbuf, N);

  zero_kernel<<<(unsigned)((aggN + TB - 1) / TB), TB, 0, stream>>>(agg, aggN);

  scatter_edges_kernel<<<(unsigned)(((long long)E * 16 + TB - 1) / TB), TB, 0, stream>>>(
      hbuf, src, dst, agg, E);

  gin_mlp1_kernel<<<(unsigned)((N + 127) / 128), TB, 0, stream>>>(
      hbuf, agg, eps1, W1a, bb1a,
      g1i, be1i, m1i, v1i, g1, be1, m1, v1, hbuf, N);

  graph_accum_kernel<<<(unsigned)(((long long)N * 16 + TB - 1) / TB), TB, 0, stream>>>(
      hbuf, gids, sums, counts, N);

  readout_kernel<<<(unsigned)((G + 127) / 128), TB, 0, stream>>>(
      sums, counts, Wr1, br1, Wr2, br2, out, G);
}
